// HardSPDNorm_54528904790466
// MI455X (gfx1250) — matrix operand order repaired and checked by reading, not compiled, not run on hardware
//
#include <hip/hip_runtime.h>
#include <hip/hip_bf16.h>
#include <math.h>

typedef __bf16 bf16;
typedef __attribute__((ext_vector_type(16))) __bf16 v16bf;
typedef __attribute__((ext_vector_type(8)))  float  v8f;

#define B_   16
#define C_   512
#define S_   4096      // 64*64 spatial
#define CT_  16        // channels per workgroup (one WMMA N tile)
#define KD_  54        // 6 in-ch * 3 * 3

// ---------------------------------------------------------------------------
// Kernel 1: fused double 2x2 avg-pool (=> 4x4 avg) and double 2x2 max-pool
// img_p/mask: (B,3,256,256) -> (B,3,64,64)
// ---------------------------------------------------------------------------
__global__ __launch_bounds__(256) void k_downsample(const float* __restrict__ img_p,
                                                    const float* __restrict__ mask,
                                                    float* __restrict__ img_ds,
                                                    float* __restrict__ mask_ds) {
  int idx = blockIdx.x * 256 + threadIdx.x;       // < 16*3*4096
  int x = idx & 63;
  int y = (idx >> 6) & 63;
  int c = idx >> 12;                               // combined b*3+ch, 0..47
  size_t base = ((size_t)c << 16) + (size_t)(y * 4) * 256 + x * 4;
  const float* ip = img_p + base;
  const float* mp = mask + base;
  float s = 0.0f, mx = -1e30f;
#pragma unroll
  for (int r = 0; r < 4; ++r) {
    float4 a = *(const float4*)(ip + r * 256);
    float4 m = *(const float4*)(mp + r * 256);
    s += a.x + a.y + a.z + a.w;
    mx = fmaxf(mx, fmaxf(fmaxf(m.x, m.y), fmaxf(m.z, m.w)));
  }
  img_ds[idx]  = s * 0.0625f;
  mask_ds[idx] = mx;
}

// ---------------------------------------------------------------------------
// Kernel 2: iterative dilation distance map. One WG per batch image (64x64).
// invariant: masked <=> m > 0 (initial m is exactly 0/1)
// ---------------------------------------------------------------------------
__global__ __launch_bounds__(256) void k_distance(const float* __restrict__ mask_ds,
                                                  float* __restrict__ Dh) {
  __shared__ float sm[S_];
  int b = blockIdx.x;
  int t = threadIdx.x;
  const float* src = mask_ds + (size_t)(b * 3) * S_;   // channel 0 of mask
  for (int i = t; i < S_; i += 256) sm[i] = src[i];
  __syncthreads();

  float decay = 1.0f;
  for (int it = 1; it <= 4; ++it) {
    decay *= 0.5f;                                     // 1 / K_DECAY^it
    float nv[16];
#pragma unroll
    for (int j = 0; j < 16; ++j) {
      int p = t * 16 + j;
      int y = p >> 6, x = p & 63;
      float cur = sm[p];
      float outv = cur;
      if (cur == 0.0f) {
        float ssum = 0.0f;
        for (int dy = -1; dy <= 1; ++dy) {
          int yy = y + dy; if (yy < 0 || yy > 63) continue;
          for (int dx = -1; dx <= 1; ++dx) {
            int xx = x + dx; if (xx < 0 || xx > 63) continue;
            ssum += sm[yy * 64 + xx];
          }
        }
        if (ssum > 0.0f) outv = decay;
      }
      nv[j] = outv;
    }
    __syncthreads();
#pragma unroll
    for (int j = 0; j < 16; ++j) sm[t * 16 + j] = nv[j];
    __syncthreads();
  }
  for (int i = t; i < S_; i += 256) Dh[(size_t)b * S_ + i] = sm[i];
}

// ---------------------------------------------------------------------------
// Kernel 3: per-(b,c) instance-norm stats. One WG per channel instance.
// ---------------------------------------------------------------------------
__global__ __launch_bounds__(256) void k_stats(const float* __restrict__ F_in,
                                               float* __restrict__ mu,
                                               float* __restrict__ rstd) {
  int bc = blockIdx.x;          // 0 .. B*C-1
  int t = threadIdx.x;
  const float* p = F_in + (size_t)bc * S_ + t * 16;
  float s = 0.0f, s2 = 0.0f;
#pragma unroll
  for (int i = 0; i < 4; ++i) {
    float4 v = *(const float4*)(p + i * 4);
    s  += v.x + v.y + v.z + v.w;
    s2 += v.x * v.x + v.y * v.y + v.z * v.z + v.w * v.w;
  }
  __shared__ float as[256], as2[256];
  as[t] = s; as2[t] = s2;
  __syncthreads();
  for (int off = 128; off > 0; off >>= 1) {
    if (t < off) { as[t] += as[t + off]; as2[t] += as2[t + off]; }
    __syncthreads();
  }
  if (t == 0) {
    float m = as[0] * (1.0f / 4096.0f);
    float var = as2[0] * (1.0f / 4096.0f) - m * m;
    mu[bc] = m;
    rstd[bc] = rsqrtf(var + 1e-5f);
  }
}

// ---------------------------------------------------------------------------
// Kernel 4: fused gated-conv (WMMA bf16 implicit GEMM, M=spatial, N=16ch,
// K=64 zero-padded from 54) + D_h gating + instance-norm affine.
// Grid: B * (C/CT) = 512 blocks, 256 threads (8 wave32).
// Dynamic LDS: bf16 img_cat halo [6][66][66] + f32 D_h [4096].
// D_h slab is staged with gfx1250 global_load_async_to_lds_b128 (ASYNCcnt).
// ---------------------------------------------------------------------------
__global__ __launch_bounds__(256) void k_main(
    const float* __restrict__ F_in,
    const float* __restrict__ Wgc, const float* __restrict__ bgc,
    const float* __restrict__ Wgm, const float* __restrict__ bgm,
    const float* __restrict__ Wbc, const float* __restrict__ bbc,
    const float* __restrict__ Wbm, const float* __restrict__ bbm,
    const float* __restrict__ img_ds, const float* __restrict__ mask_ds,
    const float* __restrict__ Dh,
    const float* __restrict__ mu, const float* __restrict__ rstd,
    float* __restrict__ out) {
  extern __shared__ char smem[];
  bf16*  sImg = (bf16*)smem;                         // [6][66][66] zero halo
  float* sDh  = (float*)(smem + 6 * 66 * 66 * 2);    // [4096]

  int b   = blockIdx.x >> 5;     // batch
  int ct  = blockIdx.x & 31;     // channel tile
  int tid = threadIdx.x;

  // L2 warm-up for this block's F_in slab (16ch x 16KB) -> global_prefetch_b8
  const float* slab = F_in + (size_t)(b * C_ + ct * CT_) * S_;
  for (int i = tid * 16; i < CT_ * S_; i += 256 * 64)
    __builtin_prefetch(slab + i, 0, 1);

  // async copy of D_h slab (16KB) straight into LDS, no VGPR round-trip:
  // global_load_async_to_lds_b128 vdst(lds-addr), vaddr(offset), saddr(base)
  {
    const float* DhB = Dh + (size_t)b * S_;
    unsigned lds_base = (unsigned)(unsigned long long)(void*)sDh;
    for (int c = tid; c < S_ / 4; c += 256) {          // 1024 x 16B chunks
      unsigned lds_addr = lds_base + c * 16;
      unsigned goff     = c * 16;
      asm volatile("global_load_async_to_lds_b128 %0, %1, %2"
                   :: "v"(lds_addr), "v"(goff), "s"(DhB)
                   : "memory");
    }
    asm volatile("s_wait_asynccnt 0" ::: "memory");
  }

  // stage img_cat (6 ch, zero-padded halo) as bf16
  for (int i = tid; i < 6 * 66 * 66; i += 256) sImg[i] = (bf16)0.0f;
  __syncthreads();
  for (int i = tid; i < 6 * S_; i += 256) {
    int ic = i >> 12;
    int p = i & (S_ - 1);
    int y = p >> 6, x = p & 63;
    float v = (ic < 3) ? img_ds[(size_t)(b * 3 + ic) * S_ + p]
                       : mask_ds[(size_t)(b * 3 + ic - 3) * S_ + p];
    sImg[(ic * 66 + (y + 1)) * 66 + (x + 1)] = (bf16)v;
  }
  __syncthreads();

  int lane = tid & 31;
  int wave = tid >> 5;
  int N = lane & 15;
  int cg = ct * CT_ + N;                 // this lane's output channel
  int khalfB = (lane < 16) ? 0 : 8;      // B frag: interleaved K per ISA layout
  int khalfA = (lane < 16) ? 0 : 8;      // A frag: interleaved K per ISA layout

  // B fragments: 4 weight sets x 2 K-chunks, resident in VGPRs
  auto loadB = [&](const float* W, int chunk) {
    v16bf r;
#pragma unroll
    for (int e = 0; e < 16; ++e) {
      int k = chunk * 32 + ((e & 8) ? 16 : 0) + khalfB + (e & 7);   // (ic*9 + dy*3 + dx) flat == weight idx
      float w = (k < KD_) ? W[(size_t)cg * KD_ + k] : 0.0f;
      r[e] = (bf16)w;
    }
    return r;
  };
  v16bf Bgc0 = loadB(Wgc, 0), Bgc1 = loadB(Wgc, 1);
  v16bf Bgm0 = loadB(Wgm, 0), Bgm1 = loadB(Wgm, 1);
  v16bf Bbc0 = loadB(Wbc, 0), Bbc1 = loadB(Wbc, 1);
  v16bf Bbm0 = loadB(Wbm, 0), Bbm1 = loadB(Wbm, 1);

  float l_bgc = bgc[cg], l_bgm = bgm[cg], l_bbc = bbc[cg], l_bbm = bbm[cg];
  float l_mu = mu[b * C_ + cg], l_rs = rstd[b * C_ + cg];

  // per-lane A-fragment LDS offsets (k -> ic,dy,dx), -1 => zero pad
  int ofsA[2][16];
#pragma unroll
  for (int ch = 0; ch < 2; ++ch)
#pragma unroll
    for (int e = 0; e < 16; ++e) {
      int k = ch * 32 + ((e & 8) ? 16 : 0) + khalfA + (e & 7);
      if (k < KD_) {
        int ic = k / 9, r9 = k % 9;
        ofsA[ch][e] = ic * (66 * 66) + (r9 / 3) * 66 + (r9 % 3);
      } else {
        ofsA[ch][e] = -1;
      }
    }

  int Ma = lane & 15;                    // A-row (spatial within tile)
  const v8f vzero = {0.f, 0.f, 0.f, 0.f, 0.f, 0.f, 0.f, 0.f};

  // each wave handles 32 contiguous 16-position spatial tiles
  for (int i = 0; i < 32; ++i) {
    int tt = wave * 32 + i;
    int m0 = tt * 16;                    // tile spans one row segment (y const)
    int y = m0 >> 6;
    int x = (m0 & 63) + Ma;
    int p0 = y * 66 + x;                 // halo-space base (dy,dx add the +1)

    v16bf A0, A1;
#pragma unroll
    for (int e = 0; e < 16; ++e) {
      A0[e] = (ofsA[0][e] >= 0) ? sImg[ofsA[0][e] + p0] : (bf16)0.0f;
      A1[e] = (ofsA[1][e] >= 0) ? sImg[ofsA[1][e] + p0] : (bf16)0.0f;
    }

    v8f aGC = __builtin_amdgcn_wmma_f32_16x16x32_bf16(false, A0, false, Bgc0, (short)0, vzero, false, false);
    aGC     = __builtin_amdgcn_wmma_f32_16x16x32_bf16(false, A1, false, Bgc1, (short)0, aGC,   false, false);
    v8f aGM = __builtin_amdgcn_wmma_f32_16x16x32_bf16(false, A0, false, Bgm0, (short)0, vzero, false, false);
    aGM     = __builtin_amdgcn_wmma_f32_16x16x32_bf16(false, A1, false, Bgm1, (short)0, aGM,   false, false);
    v8f aBC = __builtin_amdgcn_wmma_f32_16x16x32_bf16(false, A0, false, Bbc0, (short)0, vzero, false, false);
    aBC     = __builtin_amdgcn_wmma_f32_16x16x32_bf16(false, A1, false, Bbc1, (short)0, aBC,   false, false);
    v8f aBM = __builtin_amdgcn_wmma_f32_16x16x32_bf16(false, A0, false, Bbm0, (short)0, vzero, false, false);
    aBM     = __builtin_amdgcn_wmma_f32_16x16x32_bf16(false, A1, false, Bbm1, (short)0, aBM,   false, false);

    // epilogue: lane covers channel cg, spatial m = mbase + r (r = 0..7)
    int mbase = m0 + ((lane < 16) ? 0 : 8);
    size_t gbase = (size_t)(b * C_ + cg) * S_ + mbase;
    float4 f0 = *(const float4*)(F_in + gbase);
    float4 f1 = *(const float4*)(F_in + gbase + 4);
    float fv[8] = {f0.x, f0.y, f0.z, f0.w, f1.x, f1.y, f1.z, f1.w};
    float ov[8];
#pragma unroll
    for (int r = 0; r < 8; ++r) {
      float d = sDh[mbase + r];
      float h = aGC[r] + l_bgc;
      float g = aGM[r] + l_bgm;
      float gamma = ((h > 0.0f) ? h : 0.2f * h) *
                    __builtin_amdgcn_rcpf(1.0f + __expf(-g)) * d;
      float hb = aBC[r] + l_bbc;
      float gb = aBM[r] + l_bbm;
      float beta = ((hb > 0.0f) ? hb : 0.2f * hb) *
                   __builtin_amdgcn_rcpf(1.0f + __expf(-gb)) * d;
      ov[r] = (fv[r] - l_mu) * l_rs * (gamma + 1.0f) + beta;
    }
    float4 o0 = {ov[0], ov[1], ov[2], ov[3]};
    float4 o1 = {ov[4], ov[5], ov[6], ov[7]};
    *(float4*)(out + gbase)     = o0;
    *(float4*)(out + gbase + 4) = o1;
  }
}

// ---------------------------------------------------------------------------
extern "C" void kernel_launch(void* const* d_in, const int* in_sizes, int n_in,
                              void* d_out, int out_size, void* d_ws, size_t ws_size,
                              hipStream_t stream) {
  const float* F_in  = (const float*)d_in[0];
  const float* img_p = (const float*)d_in[1];
  const float* maskp = (const float*)d_in[2];
  const float* Wgc = (const float*)d_in[3];
  const float* bgc = (const float*)d_in[4];
  const float* Wgm = (const float*)d_in[5];
  const float* bgm = (const float*)d_in[6];
  const float* Wbc = (const float*)d_in[7];
  const float* bbc = (const float*)d_in[8];
  const float* Wbm = (const float*)d_in[9];
  const float* bbm = (const float*)d_in[10];
  // d_in[11] = n_ds (== 2, baked into k_downsample)
  float* out = (float*)d_out;

  float* ws      = (float*)d_ws;
  float* img_ds  = ws;                      // B*3*S   = 196608
  float* mask_ds = ws + 196608;             // B*3*S   = 196608
  float* Dh      = ws + 2 * 196608;         // B*S     = 65536
  float* mu      = Dh + (size_t)B_ * S_;    // B*C     = 8192
  float* rstd    = mu + B_ * C_;            // B*C     = 8192

  k_downsample<<<768, 256, 0, stream>>>(img_p, maskp, img_ds, mask_ds);
  k_distance<<<B_, 256, 0, stream>>>(mask_ds, Dh);
  k_stats<<<B_ * C_, 256, 0, stream>>>(F_in, mu, rstd);

  size_t shmem = (size_t)6 * 66 * 66 * 2 + (size_t)S_ * 4;  // 68656 B
  k_main<<<B_ * (C_ / CT_), 256, shmem, stream>>>(
      F_in, Wgc, bgc, Wgm, bgm, Wbc, bbc, Wbm, bbm,
      img_ds, mask_ds, Dh, mu, rstd, out);
}
